// TransformerMoEBlock_87531433493249
// MI455X (gfx1250) — matrix operand order repaired and checked by reading, not compiled, not run on hardware
//
#include <hip/hip_runtime.h>
#include <hip/hip_bf16.h>

#define N_TOK  8192
#define D_DIM  512
#define FF_DIM 2048
#define E_NUM  8
#define EPS_W  0.01f
#define NCHUNK (FF_DIM / 32)

typedef __attribute__((ext_vector_type(16))) __bf16         v16bf;
typedef __attribute__((ext_vector_type(8)))  float          v8f;
typedef __attribute__((ext_vector_type(16))) unsigned short ushort16;
typedef __attribute__((ext_vector_type(8)))  unsigned short ushort8;

union U16frag {
    ushort8        h[2];
    unsigned short s[16];
    ushort16       u;
    v16bf          v;
};

static __device__ __forceinline__ unsigned short f2bf(float f) {
    __hip_bfloat16 h = __float2bfloat16(f);
    union { __hip_bfloat16 h; unsigned short u; } cv;
    cv.h = h;
    return cv.u;
}

// Generic pointer to LDS: low 32 bits of the aperture-form address are the
// wave-relative LDS byte offset (ISA 10.2 aperture mapping).
static __device__ __forceinline__ unsigned lds_off(const void* p) {
    return (unsigned)(unsigned long long)p;
}

// One 16-byte async Global->LDS copy (ASYNCcnt-tracked), GVS addressing:
// mem = SGPR base + VGPR 32-bit byte offset.
static __device__ __forceinline__ void async_copy_b128(
    unsigned ldsByteOff, unsigned globalByteOff, const void* base) {
    asm volatile("global_load_async_to_lds_b128 %0, %1, %2"
                 :: "v"(ldsByteOff), "v"(globalByteOff), "s"(base)
                 : "memory");
}

static __device__ __forceinline__ void wait_asynccnt0() {
    asm volatile("s_wait_asynccnt 0x0" ::: "memory");
}

// ---------------------------------------------------------------------------
// fp32 -> bf16 (grid-stride), for activations
// ---------------------------------------------------------------------------
__global__ void cvt_f32_bf16(const float* __restrict__ src,
                             unsigned short* __restrict__ dst, int n) {
    int i = blockIdx.x * blockDim.x + threadIdx.x;
    int stride = gridDim.x * blockDim.x;
    for (; i < n; i += stride) dst[i] = f2bf(src[i]);
}

// ---------------------------------------------------------------------------
// fp32 [R][C] -> bf16 [C][R] per expert (tiled transpose in LDS)
// grid: (C/32, R/32, E), block: (32, 8)
// ---------------------------------------------------------------------------
__global__ __launch_bounds__(256) void cvt_transpose_bf16(
    const float* __restrict__ in, unsigned short* __restrict__ out,
    int R, int C) {
    __shared__ unsigned short tile[32][33];
    const float*    ine  = in  + (size_t)blockIdx.z * R * C;
    unsigned short* oute = out + (size_t)blockIdx.z * R * C;
    int c0 = blockIdx.x * 32, r0 = blockIdx.y * 32;
    for (int rr = threadIdx.y; rr < 32; rr += 8)
        tile[rr][threadIdx.x] = f2bf(ine[(size_t)(r0 + rr) * C + c0 + threadIdx.x]);
    __syncthreads();
    for (int cc = threadIdx.y; cc < 32; cc += 8)
        oute[(size_t)(c0 + cc) * R + r0 + threadIdx.x] = tile[threadIdx.x][cc];
}

// ---------------------------------------------------------------------------
// Gating: logits -> softmax -> top-2 -> renorm; build per-expert token lists
// ---------------------------------------------------------------------------
__global__ __launch_bounds__(256) void gate_kernel(
    const float* __restrict__ x, const float* __restrict__ Wg,
    const float* __restrict__ bg, int* __restrict__ cnt,
    int* __restrict__ listTok, float* __restrict__ listW) {
    __shared__ float WgL[D_DIM * E_NUM];
    __shared__ float bgL[E_NUM];
    int tid = threadIdx.x;
    for (int i = tid; i < D_DIM * E_NUM; i += 256) WgL[i] = Wg[i];
    if (tid < E_NUM) bgL[tid] = bg[tid];
    __syncthreads();

    int n = blockIdx.x * 256 + tid;
    const float* xr = x + (size_t)n * D_DIM;

    float acc[E_NUM];
#pragma unroll
    for (int e = 0; e < E_NUM; e++) acc[e] = bgL[e];
    for (int d = 0; d < D_DIM; d++) {
        float xv = xr[d];
#pragma unroll
        for (int e = 0; e < E_NUM; e++)
            acc[e] = fmaf(xv, WgL[d * E_NUM + e], acc[e]);
    }
    float mx = acc[0];
#pragma unroll
    for (int e = 1; e < E_NUM; e++) mx = fmaxf(mx, acc[e]);
    float g[E_NUM], sum = 0.f;
#pragma unroll
    for (int e = 0; e < E_NUM; e++) { g[e] = __expf(acc[e] - mx); sum += g[e]; }
    int i0 = 0; float v0 = g[0];
#pragma unroll
    for (int e = 1; e < E_NUM; e++) if (g[e] > v0) { v0 = g[e]; i0 = e; }
    int i1 = (i0 == 0) ? 1 : 0; float v1 = g[i1];
#pragma unroll
    for (int e = 0; e < E_NUM; e++)
        if (e != i0 && g[e] > v1) { v1 = g[e]; i1 = e; }
    v0 /= sum; v1 /= sum;
    float den = v0 + v1 + EPS_W;
    float w0 = v0 / den, w1 = v1 / den;

    int s0 = atomicAdd(&cnt[i0], 1);
    listTok[i0 * N_TOK + s0] = n;
    listW  [i0 * N_TOK + s0] = w0;
    int s1 = atomicAdd(&cnt[i1], 1);
    listTok[i1 * N_TOK + s1] = n;
    listW  [i1 * N_TOK + s1] = w1;
}

// ---------------------------------------------------------------------------
// Expert FFN: 64-token tiles, FF chunked by 32, fused bf16-WMMA GEMM pair.
// Double-buffered async Global->LDS weight staging (software pipeline).
// w1t layout: [E][FF][D] (transposed), w2t layout: [E][D][FF] (transposed).
// ---------------------------------------------------------------------------
__global__ __launch_bounds__(256) void ffn_kernel(
    const unsigned short* __restrict__ xh,
    const unsigned short* __restrict__ w1t,
    const unsigned short* __restrict__ w2t,
    const float* __restrict__ b1, const float* __restrict__ b2,
    const int* __restrict__ cnt, const int* __restrict__ listTok,
    const float* __restrict__ listW, float* __restrict__ out) {
    __shared__ __align__(16) unsigned short Xs  [64 * 512];     // tokens x D
    __shared__ __align__(16) unsigned short W1ts[2][32 * 512];  // ff-local x D
    __shared__ __align__(16) unsigned short W2ts[2][512 * 32];  // d x ff-local
    __shared__ __align__(16) unsigned short Hs  [64 * 32];      // tokens x ff-local
    __shared__ float b2s[D_DIM];
    __shared__ float b1s[2][32];
    __shared__ int   toksL[64];
    __shared__ float wtsL[64];

    int e    = blockIdx.x;
    int m0   = blockIdx.y * 64;
    int ntok = cnt[e];
    if (m0 >= ntok) return;

    int tid  = threadIdx.x;
    int lane = tid & 31;
    int wave = tid >> 5;

    if (tid < 64) {
        int r = m0 + tid;
        if (r < ntok) { toksL[tid] = listTok[e * N_TOK + r]; wtsL[tid] = listW[e * N_TOK + r]; }
        else          { toksL[tid] = 0;                      wtsL[tid] = 0.0f; }
    }
    for (int i = tid; i < D_DIM; i += 256) b2s[i] = b2[e * D_DIM + i];
    __syncthreads();

    const unsigned short* w1e = w1t + (size_t)e * FF_DIM * D_DIM;
    const unsigned short* w2e = w2t + (size_t)e * D_DIM * FF_DIM;

    // async stage of one FF chunk into buffer (c & 1)
    auto stage_async = [&](int c) {
        int buf = c & 1;
        int f0  = c * 32;
        // W1t chunk: 32 ff-rows x 512 d (1 KB contiguous per row)
        for (int i = tid; i < (32 * 512) / 8; i += 256) {
            int row = i >> 6;
            int col = (i & 63) * 8;
            async_copy_b128(lds_off(&W1ts[buf][row * 512 + col]),
                            (unsigned)(((f0 + row) * D_DIM + col) * 2), w1e);
        }
        // W2t chunk: 512 d-rows x 32 ff (64 B contiguous per row)
        for (int i = tid; i < (512 * 32) / 8; i += 256) {
            int row = i >> 2;
            int col = (i & 3) * 8;
            async_copy_b128(lds_off(&W2ts[buf][row * 32 + col]),
                            (unsigned)((row * FF_DIM + f0 + col) * 2), w2e);
        }
        if (tid < 32) b1s[buf][tid] = b1[e * FF_DIM + f0 + tid];
    };

    // prologue: async-gather 64 routed token rows + stage chunk 0
    {
        int r   = tid >> 2;   // 4 threads per row
        int seg = tid & 3;    // 256 B each
        unsigned gbase = (unsigned)((toksL[r] * D_DIM + seg * 128) * 2);
        unsigned lbase = lds_off(&Xs[r * 512 + seg * 128]);
#pragma unroll
        for (int q = 0; q < 16; q++)
            async_copy_b128(lbase + q * 16, gbase + q * 16, xh);
    }
    stage_async(0);

    v8f acc[16];
#pragma unroll
    for (int j = 0; j < 16; j++) acc[j] = (v8f)(0.0f);
    int ry    = wave & 3;          // token row tile (4 of them)
    int cbase = (wave >> 2) * 16;  // 16 of 32 col tiles

    int koffA = (lane < 16) ? 0 : 8;   // A-frag K sub-offset
    int koffB = (lane < 16) ? 0 : 16;  // B-frag K sub-offset
    int nlane = lane & 15;

    for (int c = 0; c < NCHUNK; c++) {
        int buf = c & 1;
        wait_asynccnt0();   // this chunk's async copies (and X gather) done
        __syncthreads();

        // pipeline: kick off next chunk into the other buffer
        if (c + 1 < NCHUNK) stage_async(c + 1);
        // pull chunk c+2 toward L2 while the async units fill c+1
        if (c + 2 < NCHUNK) {
            const unsigned short* p1 = w1e + (size_t)(c + 2) * 32 * D_DIM;
            if (tid < 128) __builtin_prefetch(p1 + (size_t)tid * 256, 0, 1);
            const unsigned short* p2 = w2e + (c + 2) * 32;
            __builtin_prefetch(p2 + (size_t)(tid * 2)     * FF_DIM, 0, 1);
            __builtin_prefetch(p2 + (size_t)(tid * 2 + 1) * FF_DIM, 0, 1);
        }

        // ---- GEMM 1: H(64x32) = relu(X(64x512) @ W1chunk^T) --------------
        {
            int hr = wave & 3;       // token row tile
            int hc = wave >> 2;      // ff col tile (2 of them)
            v8f hcacc = (v8f)(0.0f);
            int mrow = hr * 16 + nlane;
            int nB   = hc * 16 + nlane;  // ff-local column
            const unsigned short* w1b = &W1ts[buf][0];
#pragma unroll
            for (int kk = 0; kk < 16; kk++) {
                int kb = kk * 32;
                U16frag A, B;
                A.h[0] = *(const ushort8*)&Xs[mrow * 512 + kb + koffA];
                A.h[1] = *(const ushort8*)&Xs[mrow * 512 + kb + 16 + koffA];
                B.h[0] = *(const ushort8*)&w1b[nB * 512 + kb + koffA];
                B.h[1] = *(const ushort8*)&w1b[nB * 512 + kb + 16 + koffA];
                hcacc = __builtin_amdgcn_wmma_f32_16x16x32_bf16(
                            false, A.v, false, B.v, (short)0, hcacc, false, false);
            }
            float bias = b1s[buf][hc * 16 + nlane];
#pragma unroll
            for (int r = 0; r < 8; r++) {
                int   mloc = hr * 16 + r + ((lane < 16) ? 0 : 8);
                float hv   = hcacc[r] + bias;
                hv = hv > 0.f ? hv : 0.f;
                Hs[mloc * 32 + hc * 16 + nlane] = f2bf(hv);
            }
        }
        __syncthreads();

        // ---- GEMM 2: Y(64x512) += H(64x32) @ W2chunk^T --------------------
        {
            int m = ry * 16 + nlane;
            U16frag A;
            A.h[0] = *(const ushort8*)&Hs[m * 32 + koffA];
            A.h[1] = *(const ushort8*)&Hs[m * 32 + 16 + koffA];
            const unsigned short* w2b = &W2ts[buf][0];
#pragma unroll
            for (int j = 0; j < 16; j++) {
                int n = (cbase + j) * 16 + nlane;   // output d column
                U16frag B;
                B.h[0] = *(const ushort8*)&w2b[n * 32 + koffA];
                B.h[1] = *(const ushort8*)&w2b[n * 32 + 16 + koffA];
                acc[j] = __builtin_amdgcn_wmma_f32_16x16x32_bf16(
                            false, A.v, false, B.v, (short)0, acc[j], false, false);
            }
        }
    }

    // epilogue: add b2, scale by routing weight, atomic accumulate into out
#pragma unroll
    for (int j = 0; j < 16; j++) {
        int   dcol = (cbase + j) * 16 + nlane;
        float bb   = b2s[dcol];
#pragma unroll
        for (int r = 0; r < 8; r++) {
            int   mloc  = ry * 16 + r + ((lane < 16) ? 0 : 8);
            int   token = toksL[mloc];
            float wt    = wtsL[mloc];
            atomicAdd(&out[(size_t)token * D_DIM + dcol], (acc[j][r] + bb) * wt);
        }
    }
}

// ---------------------------------------------------------------------------
extern "C" void kernel_launch(void* const* d_in, const int* in_sizes, int n_in,
                              void* d_out, int out_size, void* d_ws, size_t ws_size,
                              hipStream_t stream) {
    const float* x  = (const float*)d_in[0];
    const float* Wg = (const float*)d_in[1];
    const float* bg = (const float*)d_in[2];
    const float* W1 = (const float*)d_in[3];
    const float* b1 = (const float*)d_in[4];
    const float* W2 = (const float*)d_in[5];
    const float* b2 = (const float*)d_in[6];
    float* out = (float*)d_out;

    char*  ws  = (char*)d_ws;
    size_t off = 0;
    auto carve = [&](size_t bytes) -> void* {
        void* p = ws + off;
        off += (bytes + 255) & ~(size_t)255;
        return p;
    };

    unsigned short* xh      = (unsigned short*)carve((size_t)N_TOK * D_DIM * 2);
    unsigned short* w1t     = (unsigned short*)carve((size_t)E_NUM * D_DIM * FF_DIM * 2);
    unsigned short* w2t     = (unsigned short*)carve((size_t)E_NUM * FF_DIM * D_DIM * 2);
    int*            cnt     = (int*)carve(E_NUM * 4);
    int*            listTok = (int*)carve((size_t)E_NUM * N_TOK * 4);
    float*          listW   = (float*)carve((size_t)E_NUM * N_TOK * 4);

    hipMemsetAsync(cnt, 0, E_NUM * 4, stream);
    hipMemsetAsync(d_out, 0, (size_t)out_size * sizeof(float), stream);

    cvt_f32_bf16<<<512, 256, 0, stream>>>(x, xh, N_TOK * D_DIM);

    // W1 [E][D][FF] -> w1t [E][FF][D]
    {
        dim3 g(FF_DIM / 32, D_DIM / 32, E_NUM);
        cvt_transpose_bf16<<<g, dim3(32, 8), 0, stream>>>(W1, w1t, D_DIM, FF_DIM);
    }
    // W2 [E][FF][D] -> w2t [E][D][FF]
    {
        dim3 g(D_DIM / 32, FF_DIM / 32, E_NUM);
        cvt_transpose_bf16<<<g, dim3(32, 8), 0, stream>>>(W2, w2t, FF_DIM, D_DIM);
    }

    gate_kernel<<<N_TOK / 256, 256, 0, stream>>>(x, Wg, bg, cnt, listTok, listW);

    dim3 grid(E_NUM, N_TOK / 64);
    ffn_kernel<<<grid, 256, 0, stream>>>(xh, w1t, w2t, b1, b2,
                                         cnt, listTok, listW, out);
}
